// CAutomaton_87651692577548
// MI455X (gfx1250) — matrix operand order repaired and checked by reading, not compiled, not run on hardware
//
#include <hip/hip_runtime.h>
#include <stdint.h>

typedef __attribute__((ext_vector_type(16))) _Float16 v16h;
typedef __attribute__((ext_vector_type(8)))  float    v8f;

#define CIN   12
#define CPCH  48
#define CHID  96
#define KDIM  108            // 9 taps * 12 channels, K order: tap*12 + c
#define NFRAG_WC 24          // 4 k-chunks * 6 m-tiles
#define NFRAG_W2 3
#define WS_WORDS ((NFRAG_WC + NFRAG_W2) * 256)   // 6912 u32 = 27648 B
#define HH 384
#define WW 384

// A-fragment K offset for 16-bit A 16x32, contiguous order:
// lanes<16: v0-7 -> K=0..15 ; lanes>=16: K=16..31
__device__ __forceinline__ int a_klo(int v, int lane) {
    return 2 * v + (lane >= 16 ? 16 : 0);
}

// u32-word offset into the xs tile for im2col index K, relative to the
// lane's pixel base ((wv*18+N)*12)/2.  Derivation: K = (dy*3+dx)*12 + c,
// xs element offset = (dy*18+dx)*12 + c = K + dy*180  ->  /2 = K/2 + 90*dy.
__host__ __device__ constexpr int o32_of(int K) {
    return K / 2 + 90 * (K / 36);
}

// ---------------- setup: fold w1 @ w_perc, pre-swizzle f16 A-fragments ------
__global__ __launch_bounds__(256)
void nca_setup(const float* __restrict__ w_perc,   // [48][12][3][3]
               const float* __restrict__ w1,       // [96][48]
               const float* __restrict__ w2,       // [12][96]
               uint32_t* __restrict__ wsf) {
    __shared__ float WC[CHID][KDIM];
    const int t = threadIdx.x;
    for (int i = t; i < CHID * KDIM; i += 256) {
        int o = i / KDIM, k = i % KDIM;
        int g = k / 12, c = k % 12;            // g = tap (dy*3+dx)
        float s = 0.f;
        #pragma unroll 8
        for (int cp = 0; cp < CPCH; ++cp)
            s += w1[o * CPCH + cp] * w_perc[(cp * CIN + c) * 9 + g];
        WC[o][k] = s;
    }
    __syncthreads();
    for (int i = t; i < WS_WORDS; i += 256) {
        int f = i >> 8;              // fragment id
        int r = i & 255;
        int lane = r >> 3, v = r & 7;
        int klo = a_klo(v, lane);
        union { _Float16 h[2]; uint32_t u; } pk; pk.u = 0u;
        if (f < NFRAG_WC) {
            int kc = f / 6, mt = f % 6;
            int K = kc * 32 + klo;                    // even
            int M = mt * 16 + (lane & 15);
            if (K < KDIM) {                           // K even, 108 even -> K+1 also valid
                pk.h[0] = (_Float16)WC[M][K];
                pk.h[1] = (_Float16)WC[M][K + 1];
            }
        } else {
            int kc = f - NFRAG_WC;
            int K = kc * 32 + klo;                    // < 96 always
            int M = lane & 15;
            if (M < CIN) {
                pk.h[0] = (_Float16)w2[M * CHID + K];
                pk.h[1] = (_Float16)w2[M * CHID + K + 1];
            }
        }
        wsf[i] = pk.u;
    }
}

// ---------------- main fused kernel -----------------------------------------
__global__ __launch_bounds__(256)
void nca_main(const float* __restrict__ x,
              const float* __restrict__ b1,
              const float* __restrict__ b2,
              const float* __restrict__ mask,
              const uint32_t* __restrict__ wsf,
              float* __restrict__ out) {
    __shared__ __align__(32) uint32_t wfrag[WS_WORDS];        // 27648 B
    __shared__ __align__(16) _Float16 xs[10 * 18 * 12];       //  4320 B
    __shared__ __align__(16) _Float16 hbuf[8 * CHID * 16];    // 24576 B
    __shared__ float b1s[CHID];
    __shared__ float b2s[16];

    const int t  = threadIdx.x;
    const int tx = blockIdx.x, ty = blockIdx.y, b = blockIdx.z;

    // weights -> LDS
    for (int i = t; i < WS_WORDS; i += 256) wfrag[i] = wsf[i];
    if (t < CHID) b1s[t] = b1[t];
    if (t < 16)   b2s[t] = (t < CIN) ? b2[t] : 0.f;

    // x tile -> LDS f16, layout [row][col][c] (wrap-padded halo of 1)
    for (int i = t; i < 10 * 12 * 18; i += 256) {
        int r   = i / (12 * 18);
        int c   = (i / 18) % 12;
        int col = i % 18;
        int gy = (ty * 8 + r + (HH - 1)) % HH;
        int gx = (tx * 16 + col + (WW - 1)) % WW;
        xs[(r * 18 + col) * 12 + c] =
            (_Float16)x[((size_t)(b * CIN + c) * HH + gy) * WW + gx];
    }
    __syncthreads();

    const int lane = t & 31;
    const int wv   = t >> 5;          // wave id == tile row
    const int N    = lane & 15;       // WMMA column == pixel within tile row
    const int hi   = lane >> 4;       // half-wave

    const uint32_t* xs32 = (const uint32_t*)xs;
    // per-lane base (u32 words); +8 folds the hi half-wave's K+16 linear part
    const int base32h = (((wv * 18 + N) * 12) >> 1) + hi * 8;

    // ---- GEMM1: h[96][16] = relu(WC[96x108] @ im2col[108x16] + b1)
    v8f acc[6] = {};
    #pragma unroll
    for (int kc = 0; kc < 4; ++kc) {
        union { uint32_t u[8]; v16h h; } bf;
        #pragma unroll
        for (int v = 0; v < 8; ++v) {
            const int K0 = kc * 32 + 2 * v;     // compile-time after unroll
            const int K1 = K0 + 16;
            uint32_t w;
            if (K0 >= KDIM) {
                w = 0u;                                       // both halves pad
            } else if (K1 >= KDIM) {
                w = hi ? 0u : xs32[base32h + o32_of(K0)];     // hi half pad
            } else {
                const int o0 = o32_of(K0);            // rel. to base (hi=0)
                const int o1 = o32_of(K1) - 8;        // rel. to base32h (hi=1)
                if (o0 == o1) {
                    w = xs32[base32h + o0];           // pure DS immediate
                } else {
                    w = xs32[base32h + (hi ? o1 : o0)];  // 1 cndmask of literals
                }
            }
            bf.u[v] = w;
        }
        #pragma unroll
        for (int mt = 0; mt < 6; ++mt) {
            const v16h* ap = (const v16h*)&wfrag[((kc * 6 + mt) * 32 + lane) * 8];
            acc[mt] = __builtin_amdgcn_wmma_f32_16x16x32_f16(
                false, *ap, false, bf.h, (short)0, acc[mt], false, false);
        }
    }

    // ---- bias + ReLU, repack h to per-wave LDS [N][M] as f16
    uint32_t* hb32  = (uint32_t*)&hbuf[wv * CHID * 16];
    uint32_t* hbst  = hb32 + N * 48 + hi * 4;       // store base, const offsets
    #pragma unroll
    for (int mt = 0; mt < 6; ++mt) {
        #pragma unroll
        for (int v = 0; v < 8; v += 2) {
            int M = mt * 16 + v + hi * 8;           // even
            float a0 = acc[mt][v]     + b1s[M];
            float a1 = acc[mt][v + 1] + b1s[M + 1];
            a0 = a0 > 0.f ? a0 : 0.f;
            a1 = a1 > 0.f ? a1 : 0.f;
            union { _Float16 h[2]; uint32_t u; } pk;
            pk.h[0] = (_Float16)a0; pk.h[1] = (_Float16)a1;
            hbst[mt * 8 + (v >> 1)] = pk.u;
        }
    }
    asm volatile("s_wait_dscnt 0" ::: "memory");    // intra-wave LDS RAW

    // ---- GEMM2: update[12][16] = w2[12x96] @ h[96x16]  (M padded to 16)
    const uint32_t* hbld = hb32 + N * 48 + hi * 8;  // load base, const offsets
    v8f acc2 = {};
    #pragma unroll
    for (int kc = 0; kc < 3; ++kc) {
        union { uint32_t u[8]; v16h h; } bf;
        #pragma unroll
        for (int v = 0; v < 8; ++v)
            bf.u[v] = hbld[kc * 16 + v];
        const v16h* ap = (const v16h*)&wfrag[(NFRAG_WC + kc) * 256 + lane * 8];
        acc2 = __builtin_amdgcn_wmma_f32_16x16x32_f16(
            false, *ap, false, bf.h, (short)0, acc2, false, false);
    }

    // ---- masked residual store (fp32 exact residual: reload x from L2)
    const int gy = ty * 8 + wv;
    const int gx = tx * 16 + N;
    const float mbit = (mask[gy * WW + gx] > 0.5f) ? 1.f : 0.f;
    const size_t idx0 = ((size_t)(b * CIN) * HH + gy) * WW + gx;
    #pragma unroll
    for (int v = 0; v < 8; ++v) {
        int M = v + hi * 8;                         // D layout: lanes>=16 -> M+8
        if (M < CIN) {
            size_t idx = idx0 + (size_t)M * (HH * WW);
            out[idx] = x[idx] + mbit * (acc2[v] + b2s[M]);
        }
    }
}

extern "C" void kernel_launch(void* const* d_in, const int* in_sizes, int n_in,
                              void* d_out, int out_size, void* d_ws, size_t ws_size,
                              hipStream_t stream) {
    (void)in_sizes; (void)n_in; (void)out_size; (void)ws_size;
    const float* x      = (const float*)d_in[0];
    const float* w_perc = (const float*)d_in[1];
    const float* w1     = (const float*)d_in[2];
    const float* b1     = (const float*)d_in[3];
    const float* w2     = (const float*)d_in[4];
    const float* b2     = (const float*)d_in[5];
    const float* mask   = (const float*)d_in[6];
    uint32_t* wsf = (uint32_t*)d_ws;

    nca_setup<<<1, 256, 0, stream>>>(w_perc, w1, w2, wsf);

    dim3 grid(WW / 16, HH / 8, 16);
    nca_main<<<grid, 256, 0, stream>>>(x, b1, b2, mask, wsf, (float*)d_out);
}
